// rwkv7MoeBlock_65661460021709
// MI455X (gfx1250) — compile-verified, matrix operand order checked by reading, not run on hardware
//
#include <hip/hip_runtime.h>

// ---------------------------------------------------------------------------
// RWKV7 + hash-MoE block for MI455X (gfx1250, wave32, WMMA bf16).
// Heavy GEMMs: v_wmma_f32_16x16x32_bf16, double-buffered LDS tiles,
// A-tiles staged with global_load_async_to_lds_b128 (ASYNCcnt pipeline),
// fp32 weights converted to bf16 on the LDS staging path.
// Sequential WKV recurrence isolated to 32 (batch,head) blocks.
// ---------------------------------------------------------------------------

constexpr int B_  = 2;
constexpr int T_  = 1024;
constexpr int C_  = 1024;
constexpr int A_  = 1024;
constexpr int N_  = 64;
constexpr int H_  = 16;
constexpr int I_  = 3584;
constexpr int E_  = 8;
constexpr int BT_ = B_ * T_;
constexpr float LNX_EPS_ = 1e-5f * 64.0f;

typedef __attribute__((ext_vector_type(16))) __bf16 v16bf;
typedef __attribute__((ext_vector_type(8)))  float  v8f;

union FragBF {
  unsigned int u[8];
  v16bf v;
};

__device__ __forceinline__ unsigned short f32_to_bf16(float f) {
  unsigned int u = __float_as_uint(f);
  unsigned int r = (u + 0x7fffu + ((u >> 16) & 1u)) >> 16;
  return (unsigned short)r;
}

enum { EPI_NONE = 0, EPI_SILU = 1, EPI_SIGMOID = 2, EPI_TANH = 3, EPI_RELUSQ = 4 };

template <int EPI>
__device__ __forceinline__ float apply_epi(float v) {
  if (EPI == EPI_SILU)    return v / (1.0f + __expf(-v));
  if (EPI == EPI_SIGMOID) return 1.0f / (1.0f + __expf(-v));
  if (EPI == EPI_TANH)    return tanhf(v);
  if (EPI == EPI_RELUSQ)  { float t = v > 0.0f ? v : 0.0f; return t * t; }
  return v;
}

// ---------------------------------------------------------------------------
// Tiled WMMA GEMM:  Out[M,Nout] = epi( Abf16[M,K] x W + bias ) (+residual)
//   WT=true : W is [Nout,K] row-major (out = A @ W^T)    — weights fp32
//   WT=false: W is [K,Nout] row-major (out = A @ W)
//   rowIdxA : optional gather of A rows; rowIdxOut+ACC: scatter-add into Out
//   mCnt    : optional device row count (dynamic M for routed experts)
// Block: 256 threads = 8 waves; tile 128x128x32; each wave = 4x2 WMMA tiles.
// A-tiles are staged with async global->LDS copies; tiles double-buffered so
// stage(k+1) overlaps WMMA(k) with one barrier per K-step.
// ---------------------------------------------------------------------------
template <bool OUT_BF16, int EPI, bool WT, bool ACC>
__global__ __launch_bounds__(256) void gemm_bf16_wmma(
    const unsigned short* __restrict__ A, const float* __restrict__ W,
    void* __restrict__ Out,
    const float* __restrict__ bias, const float* __restrict__ residual,
    const int* __restrict__ rowIdxA, const int* __restrict__ rowIdxOut,
    const int* __restrict__ mCnt, int M, int Nout, int K) {
  __shared__ __align__(16) unsigned short As[2][128][40];
  __shared__ __align__(16) unsigned short Bs[2][128][40];

  const int Mval = mCnt ? *mCnt : M;
  const int m0 = blockIdx.x * 128;
  if (m0 >= Mval) return;
  const int n0 = blockIdx.y * 128;

  const int tid  = threadIdx.x;
  const int lane = tid & 31;
  const int wv   = tid >> 5;
  const int wm   = (wv & 1) * 64;   // wave M offset in tile
  const int wn   = (wv >> 1) * 32;  // wave N offset in tile
  const int laneHi = (lane >= 16) ? 1 : 0;
  const int lmod   = lane & 15;

  const v8f vzero = {0.f, 0.f, 0.f, 0.f, 0.f, 0.f, 0.f, 0.f};
  v8f acc[4][2];
#pragma unroll
  for (int i = 0; i < 4; ++i)
#pragma unroll
    for (int j = 0; j < 2; ++j) acc[i][j] = vzero;

  // A-tile loader mapping: each thread owns 16 bf16 (32 B) of one tile row
  const int arow = tid >> 1;
  const int acb  = (tid & 1) * 16;
  const int gmr  = m0 + arow;
  long agrow = -1;
  if (gmr < Mval) agrow = rowIdxA ? (long)rowIdxA[gmr] : (long)gmr;

  const int nKt = K / 32;

  // ---- tile stagers -------------------------------------------------------
  auto stageA = [&](int kt, int b) {
    const int kb = kt * 32;
    if (agrow >= 0) {
      // async global -> LDS copy (32 bytes = 2x b128); instruction offset is
      // applied to both the LDS and the global address per the ISA.
      const unsigned long long ga =
          (unsigned long long)(A + (size_t)agrow * K + kb + acb);
      const unsigned int la = (unsigned int)(size_t)&As[b][arow][acb];
      asm volatile(
          "global_load_async_to_lds_b128 %0, %1, off\n\t"
          "global_load_async_to_lds_b128 %0, %1, off offset:16"
          :
          : "v"(la), "v"(ga)
          : "memory");
    } else {
#pragma unroll
      for (int q = 0; q < 8; ++q) *(unsigned int*)&As[b][arow][acb + 2 * q] = 0u;
    }
  };
  auto stageW = [&](int kt, int b) {
    const int kb = kt * 32;
    if (WT) {
      const int nr = tid >> 1;
      const int cb = (tid & 1) * 16;
      const int gn = n0 + nr;
      if (gn < Nout) {
        const float* src = W + (size_t)gn * K + kb + cb;
        if (kt + 1 < nKt) __builtin_prefetch(src + 32, 0, 1);  // global_prefetch_b8
#pragma unroll
        for (int q = 0; q < 8; ++q) {
          unsigned int lo = f32_to_bf16(src[2 * q]);
          unsigned int hi = f32_to_bf16(src[2 * q + 1]);
          *(unsigned int*)&Bs[b][nr][cb + 2 * q] = lo | (hi << 16);
        }
      } else {
#pragma unroll
        for (int q = 0; q < 8; ++q) *(unsigned int*)&Bs[b][nr][cb + 2 * q] = 0u;
      }
    } else {
      const int kr   = tid >> 3;         // 0..31
      const int cseg = (tid & 7) * 16;   // 0..112
      const float* src = W + (size_t)(kb + kr) * Nout + n0 + cseg;
#pragma unroll
      for (int q = 0; q < 16; ++q) {
        float fv = (n0 + cseg + q < Nout) ? src[q] : 0.0f;
        Bs[b][cseg + q][kr] = f32_to_bf16(fv);
      }
    }
  };

  // ---- pipelined main loop ------------------------------------------------
  stageA(0, 0);
  stageW(0, 0);

  for (int kt = 0; kt < nKt; ++kt) {
    const int b = kt & 1;
    // retire this wave's async LDS writes for buffer b, then publish to WG
    asm volatile("s_wait_asynccnt 0" ::: "memory");
    __syncthreads();
    if (kt + 1 < nKt) {  // overlap: stage next tile into the other buffer
      stageA(kt + 1, b ^ 1);
      stageW(kt + 1, b ^ 1);
    }

    // ---- fragments per ISA 16-bit A 16x32 layout ----
    FragBF fa[4];
#pragma unroll
    for (int am = 0; am < 4; ++am) {
      const int row = wm + am * 16 + lmod;
#pragma unroll
      for (int j = 0; j < 8; ++j) {
        const int p = (j & 3) + ((j >= 4) ? 8 : 0) + (laneHi ? 4 : 0);
        fa[am].u[j] = *(const unsigned int*)&As[b][row][2 * p];
      }
    }
    FragBF fb[2];
#pragma unroll
    for (int bn = 0; bn < 2; ++bn) {
      const int col = wn + bn * 16 + lmod;
#pragma unroll
      for (int j = 0; j < 8; ++j) {
        const int p = (j & 3) + ((j >= 4) ? 8 : 0) + (laneHi ? 4 : 0);
        fb[bn].u[j] = *(const unsigned int*)&Bs[b][col][2 * p];
      }
    }
#pragma unroll
    for (int am = 0; am < 4; ++am)
#pragma unroll
      for (int bn = 0; bn < 2; ++bn)
        acc[am][bn] = __builtin_amdgcn_wmma_f32_16x16x32_bf16(
            false, fa[am].v, false, fb[bn].v, (short)0, acc[am][bn], false, false);
  }

  // ---- epilogue / store (C layout: VGPR r -> M = r + 8*laneHi, N = lane&15) --
#pragma unroll
  for (int am = 0; am < 4; ++am) {
#pragma unroll
    for (int bn = 0; bn < 2; ++bn) {
#pragma unroll
      for (int r = 0; r < 8; ++r) {
        const int mr = m0 + wm + am * 16 + r + (laneHi ? 8 : 0);
        const int nc = n0 + wn + bn * 16 + lmod;
        if (mr < Mval && nc < Nout) {
          float v = acc[am][bn][r];
          if (bias) v += bias[nc];
          v = apply_epi<EPI>(v);
          if (ACC) {
            const long orow = rowIdxOut ? (long)rowIdxOut[mr] : (long)mr;
            float* o = (float*)Out;
            o[(size_t)orow * Nout + nc] += v;
          } else if (OUT_BF16) {
            ((unsigned short*)Out)[(size_t)mr * Nout + nc] = f32_to_bf16(v);
          } else {
            if (residual) v += residual[(size_t)mr * Nout + nc];
            ((float*)Out)[(size_t)mr * Nout + nc] = v;
          }
        }
      }
    }
  }
}

// ---------------------------------------------------------------------------
// Block-wide helpers (256 threads, rows of 1024)
// ---------------------------------------------------------------------------
__device__ __forceinline__ float blk_sum(float v, float* red) {
  const int tid = threadIdx.x;
  red[tid] = v;
  __syncthreads();
  for (int s = 128; s > 0; s >>= 1) {
    if (tid < s) red[tid] += red[tid + s];
    __syncthreads();
  }
  float r = red[0];
  __syncthreads();
  return r;
}

__device__ __forceinline__ void ln1024(const float* __restrict__ src, float* __restrict__ dst,
                                       const float* __restrict__ g, const float* __restrict__ b,
                                       float* red, float eps) {
  const int tid = threadIdx.x;
  float s = 0.0f;
#pragma unroll
  for (int j = 0; j < 4; ++j) s += src[tid + 256 * j];
  const float m = blk_sum(s, red) * (1.0f / 1024.0f);
  float vs = 0.0f;
#pragma unroll
  for (int j = 0; j < 4; ++j) {
    float d = src[tid + 256 * j] - m;
    vs += d * d;
  }
  const float var = blk_sum(vs, red) * (1.0f / 1024.0f);
  const float inv = rsqrtf(var + eps);
#pragma unroll
  for (int j = 0; j < 4; ++j) {
    int c = tid + 256 * j;
    dst[c] = (src[c] - m) * inv * g[c] + b[c];
  }
  __syncthreads();
}

// ---------------------------------------------------------------------------
// attn_prep: pre_ln -> x1 (store), ln1 -> h; recompute h[t-1] for the shift;
// token-mix tanh(xxx@w1)@w2; emit bf16 Zw/Zk/Zv/Zr/Zg.
// ---------------------------------------------------------------------------
__global__ __launch_bounds__(256) void attn_prep_kernel(
    const float* __restrict__ hidden,
    const float* __restrict__ pre_g, const float* __restrict__ pre_b,
    const float* __restrict__ ln1_g, const float* __restrict__ ln1_b,
    const float* __restrict__ maa_x,
    const float* __restrict__ maa_w, const float* __restrict__ maa_k,
    const float* __restrict__ maa_v, const float* __restrict__ maa_r,
    const float* __restrict__ maa_g,
    const float* __restrict__ w1, const float* __restrict__ w2,
    float* __restrict__ x1,
    unsigned short* __restrict__ Zw, unsigned short* __restrict__ Zk,
    unsigned short* __restrict__ Zv, unsigned short* __restrict__ Zr,
    unsigned short* __restrict__ Zg) {
  __shared__ float s_tmp[1024], s_h[1024], s_hp[1024], s_xxx[1024];
  __shared__ float s_t5[160];
  __shared__ float s_red[256];
  const int i = blockIdx.x;
  const int t = i % T_;
  const int tid = threadIdx.x;

  ln1024(hidden + (size_t)i * C_, s_tmp, pre_g, pre_b, s_red, 1e-5f);
#pragma unroll
  for (int j = 0; j < 4; ++j) x1[(size_t)i * C_ + tid + 256 * j] = s_tmp[tid + 256 * j];
  ln1024(s_tmp, s_h, ln1_g, ln1_b, s_red, 1e-5f);

  if (t > 0) {
    ln1024(hidden + (size_t)(i - 1) * C_, s_tmp, pre_g, pre_b, s_red, 1e-5f);
    ln1024(s_tmp, s_hp, ln1_g, ln1_b, s_red, 1e-5f);
  } else {
#pragma unroll
    for (int j = 0; j < 4; ++j) s_hp[tid + 256 * j] = 0.0f;
    __syncthreads();
  }

#pragma unroll
  for (int j = 0; j < 4; ++j) {
    int c = tid + 256 * j;
    s_xxx[c] = s_h[c] + (s_hp[c] - s_h[c]) * maa_x[c];
  }
  __syncthreads();

  if (tid < 160) {
    float a = 0.0f;
    for (int c = 0; c < C_; ++c) a += s_xxx[c] * w1[(size_t)c * 160 + tid];
    s_t5[tid] = tanhf(a);
  }
  __syncthreads();

  const float* maas[5] = {maa_w, maa_k, maa_v, maa_r, maa_g};
  unsigned short* Zs[5] = {Zw, Zk, Zv, Zr, Zg};
#pragma unroll
  for (int f = 0; f < 5; ++f) {
#pragma unroll
    for (int j = 0; j < 4; ++j) {
      int c = tid + 256 * j;
      float m = 0.0f;
#pragma unroll
      for (int d = 0; d < 32; ++d) m += s_t5[f * 32 + d] * w2[(size_t)(f * 32 + d) * C_ + c];
      float xxv = s_hp[c] - s_h[c];
      float z = s_h[c] + xxv * (maas[f][c] + m);
      Zs[f][(size_t)i * C_ + c] = f32_to_bf16(z);
    }
  }
}

// ---------------------------------------------------------------------------
// WKV recurrence: one block per (batch, head); lane m owns state column s[:,m]
// ---------------------------------------------------------------------------
__global__ __launch_bounds__(64) void wkv_kernel(
    const float* __restrict__ r, const float* __restrict__ k,
    const float* __restrict__ v, const float* __restrict__ wraw,
    const float* __restrict__ u, float* __restrict__ out) {
  __shared__ float rr[64], kk[64], vv[64], ww[64], uu[64];
  const int bh = blockIdx.x;
  const int b = bh / H_;
  const int hh = bh % H_;
  const int tid = threadIdx.x;
  uu[tid] = u[hh * 64 + tid];
  float s[64];
#pragma unroll
  for (int n = 0; n < 64; ++n) s[n] = 0.0f;

  for (int t = 0; t < T_; ++t) {
    const size_t base = (size_t)(b * T_ + t) * A_ + hh * 64;
    rr[tid] = r[base + tid];
    kk[tid] = k[base + tid];
    vv[tid] = v[base + tid];
    ww[tid] = __expf(-__expf(wraw[base + tid]));
    __syncthreads();
    float c1 = 0.0f;
#pragma unroll
    for (int n = 0; n < 64; ++n) c1 += rr[n] * uu[n] * kk[n];
    const float vm = vv[tid];
    float o = vm * c1;
#pragma unroll
    for (int n = 0; n < 64; ++n) {
      o += rr[n] * s[n];
      s[n] = kk[n] * vm + ww[n] * s[n];
    }
    out[base + tid] = o;
    __syncthreads();
  }
}

// ---------------------------------------------------------------------------
// group-norm(H groups of 64) fused with silu-gate multiply -> bf16
// ---------------------------------------------------------------------------
__global__ __launch_bounds__(256) void gn_gate_kernel(
    const float* __restrict__ wkvo, const float* __restrict__ gpre,
    const float* __restrict__ lnx_g, const float* __restrict__ lnx_b,
    unsigned short* __restrict__ Gg) {
  __shared__ float a[1024];
  __shared__ float red[256];
  __shared__ float gm[16], gv[16];
  const int i = blockIdx.x;
  const int tid = threadIdx.x;
  const int grp = tid >> 4, sub = tid & 15;
#pragma unroll
  for (int j = 0; j < 4; ++j) a[tid + 256 * j] = wkvo[(size_t)i * A_ + tid + 256 * j];
  __syncthreads();
  float ps = 0.0f;
#pragma unroll
  for (int q = 0; q < 4; ++q) ps += a[grp * 64 + sub * 4 + q];
  red[tid] = ps;
  __syncthreads();
  if (sub == 0) {
    float s = 0.0f;
    for (int kq = 0; kq < 16; ++kq) s += red[grp * 16 + kq];
    gm[grp] = s * (1.0f / 64.0f);
  }
  __syncthreads();
  const float m = gm[grp];
  float vs = 0.0f;
#pragma unroll
  for (int q = 0; q < 4; ++q) {
    float d = a[grp * 64 + sub * 4 + q] - m;
    vs += d * d;
  }
  red[tid] = vs;
  __syncthreads();
  if (sub == 0) {
    float s = 0.0f;
    for (int kq = 0; kq < 16; ++kq) s += red[grp * 16 + kq];
    gv[grp] = s * (1.0f / 64.0f);
  }
  __syncthreads();
  const float inv = rsqrtf(gv[grp] + LNX_EPS_);
#pragma unroll
  for (int q = 0; q < 4; ++q) {
    int c = grp * 64 + sub * 4 + q;
    float nv = (a[c] - m) * inv * lnx_g[c] + lnx_b[c];
    float gate = gpre[(size_t)i * A_ + c];  // already silu'd by GEMM epilogue
    Gg[(size_t)i * A_ + c] = f32_to_bf16(gate * nv);
  }
}

// ---------------------------------------------------------------------------
// ffn_prep: ln2 + time-shift mixing -> bf16 Zr2/Zk2
// ---------------------------------------------------------------------------
__global__ __launch_bounds__(256) void ffn_prep_kernel(
    const float* __restrict__ x2,
    const float* __restrict__ ln2_g, const float* __restrict__ ln2_b,
    const float* __restrict__ maa_k, const float* __restrict__ maa_r,
    unsigned short* __restrict__ Zk2, unsigned short* __restrict__ Zr2) {
  __shared__ float s_h[1024], s_hp[1024];
  __shared__ float s_red[256];
  const int i = blockIdx.x;
  const int t = i % T_;
  const int tid = threadIdx.x;
  ln1024(x2 + (size_t)i * C_, s_h, ln2_g, ln2_b, s_red, 1e-5f);
  if (t > 0) {
    ln1024(x2 + (size_t)(i - 1) * C_, s_hp, ln2_g, ln2_b, s_red, 1e-5f);
  } else {
#pragma unroll
    for (int j = 0; j < 4; ++j) s_hp[tid + 256 * j] = 0.0f;
    __syncthreads();
  }
#pragma unroll
  for (int j = 0; j < 4; ++j) {
    int c = tid + 256 * j;
    float d = s_hp[c] - s_h[c];
    Zr2[(size_t)i * C_ + c] = f32_to_bf16(s_h[c] + d * maa_r[c]);
    Zk2[(size_t)i * C_ + c] = f32_to_bf16(s_h[c] + d * maa_k[c]);
  }
}

// ---------------------------------------------------------------------------
// hash routing: counting + token lists
// ---------------------------------------------------------------------------
__global__ void route_zero_kernel(int* counts) {
  if (threadIdx.x < E_) counts[threadIdx.x] = 0;
}
__global__ void route_build_kernel(const int* __restrict__ ids, int* counts,
                                   int* __restrict__ lists) {
  const int i = blockIdx.x * 256 + threadIdx.x;
  if (i < BT_) {
    int e = (ids[i] * 5099) % E_;
    int pos = atomicAdd(&counts[e], 1);
    lists[e * BT_ + pos] = i;
  }
}

__global__ void final_kernel(const float* __restrict__ x2, const float* __restrict__ recept,
                             const float* __restrict__ val, float* __restrict__ out) {
  const size_t idx = (size_t)blockIdx.x * 256 + threadIdx.x;
  if (idx < (size_t)BT_ * C_) out[idx] = x2[idx] + recept[idx] * val[idx];
}

// ---------------------------------------------------------------------------
extern "C" void kernel_launch(void* const* d_in, const int* in_sizes, int n_in,
                              void* d_out, int out_size, void* d_ws, size_t ws_size,
                              hipStream_t stream) {
  (void)in_sizes; (void)n_in; (void)out_size; (void)ws_size;
  const float* hidden   = (const float*)d_in[0];
  const int*   ids      = (const int*)d_in[1];
  const float* pre_g    = (const float*)d_in[2];
  const float* pre_b    = (const float*)d_in[3];
  const float* ln1_g    = (const float*)d_in[4];
  const float* ln1_b    = (const float*)d_in[5];
  const float* ln2_g    = (const float*)d_in[6];
  const float* ln2_b    = (const float*)d_in[7];
  const float* maa_x    = (const float*)d_in[8];
  const float* maa_w    = (const float*)d_in[9];
  const float* maa_k    = (const float*)d_in[10];
  const float* maa_v    = (const float*)d_in[11];
  const float* maa_r    = (const float*)d_in[12];
  const float* maa_g    = (const float*)d_in[13];
  const float* maa_w1   = (const float*)d_in[14];
  const float* maa_w2   = (const float*)d_in[15];
  const float* att_td   = (const float*)d_in[16];
  const float* td_w1    = (const float*)d_in[17];
  const float* td_w2    = (const float*)d_in[18];
  const float* att_u    = (const float*)d_in[19];
  const float* att_wr   = (const float*)d_in[20];
  const float* att_wk   = (const float*)d_in[21];
  const float* att_wv   = (const float*)d_in[22];
  const float* att_wg   = (const float*)d_in[23];
  const float* att_wo   = (const float*)d_in[24];
  const float* lnx_g    = (const float*)d_in[25];
  const float* lnx_b    = (const float*)d_in[26];
  const float* fmaa_k   = (const float*)d_in[27];
  const float* fmaa_r   = (const float*)d_in[28];
  const float* ffn_wr   = (const float*)d_in[29];
  const float* sh_wk    = (const float*)d_in[30];
  const float* sh_wv    = (const float*)d_in[31];
  const float* ex_wk    = (const float*)d_in[32];
  const float* ex_wv    = (const float*)d_in[33];
  float* out = (float*)d_out;

  // ---- workspace carve-up ----
  char* wsp = (char*)d_ws;
  auto alloc = [&](size_t bytes) {
    void* p = (void*)wsp;
    wsp += (bytes + 255) & ~(size_t)255;
    return p;
  };
  float* x1             = (float*)alloc((size_t)BT_ * C_ * 4);
  unsigned short* Zw    = (unsigned short*)alloc((size_t)BT_ * C_ * 2);
  unsigned short* Zk    = (unsigned short*)alloc((size_t)BT_ * C_ * 2);
  unsigned short* Zv    = (unsigned short*)alloc((size_t)BT_ * C_ * 2);
  unsigned short* Zr    = (unsigned short*)alloc((size_t)BT_ * C_ * 2);
  unsigned short* Zg    = (unsigned short*)alloc((size_t)BT_ * C_ * 2);
  float* rbuf           = (float*)alloc((size_t)BT_ * A_ * 4);
  float* kbuf           = (float*)alloc((size_t)BT_ * A_ * 4);
  float* vbuf           = (float*)alloc((size_t)BT_ * A_ * 4);
  float* gbuf           = (float*)alloc((size_t)BT_ * A_ * 4);
  unsigned short* tdmid = (unsigned short*)alloc((size_t)BT_ * 64 * 2);
  float* wraw           = (float*)alloc((size_t)BT_ * A_ * 4);
  float* wkvo           = (float*)alloc((size_t)BT_ * A_ * 4);
  unsigned short* Gg    = (unsigned short*)alloc((size_t)BT_ * A_ * 2);
  float* x2             = (float*)alloc((size_t)BT_ * C_ * 4);
  unsigned short* Zr2   = (unsigned short*)alloc((size_t)BT_ * C_ * 2);
  unsigned short* Zk2   = (unsigned short*)alloc((size_t)BT_ * C_ * 2);
  float* recept         = (float*)alloc((size_t)BT_ * C_ * 4);
  unsigned short* Hk    = (unsigned short*)alloc((size_t)BT_ * I_ * 2);
  float* val            = (float*)alloc((size_t)BT_ * C_ * 4);
  int* counts           = (int*)alloc(E_ * 4);
  int* lists            = (int*)alloc((size_t)E_ * BT_ * 4);

  const dim3 blk(256);
  auto g2 = [](int M, int N) { return dim3((M + 127) / 128, (N + 127) / 128); };

  // ---- attention half ----
  attn_prep_kernel<<<BT_, blk, 0, stream>>>(hidden, pre_g, pre_b, ln1_g, ln1_b, maa_x,
                                            maa_w, maa_k, maa_v, maa_r, maa_g,
                                            maa_w1, maa_w2, x1, Zw, Zk, Zv, Zr, Zg);

  gemm_bf16_wmma<false, EPI_NONE, true, false><<<g2(BT_, A_), blk, 0, stream>>>(
      Zr, att_wr, rbuf, nullptr, nullptr, nullptr, nullptr, nullptr, BT_, A_, C_);
  gemm_bf16_wmma<false, EPI_NONE, true, false><<<g2(BT_, A_), blk, 0, stream>>>(
      Zk, att_wk, kbuf, nullptr, nullptr, nullptr, nullptr, nullptr, BT_, A_, C_);
  gemm_bf16_wmma<false, EPI_NONE, true, false><<<g2(BT_, A_), blk, 0, stream>>>(
      Zv, att_wv, vbuf, nullptr, nullptr, nullptr, nullptr, nullptr, BT_, A_, C_);
  gemm_bf16_wmma<false, EPI_SILU, true, false><<<g2(BT_, A_), blk, 0, stream>>>(
      Zg, att_wg, gbuf, nullptr, nullptr, nullptr, nullptr, nullptr, BT_, A_, C_);

  // td = tanh(zw @ td_w1) @ td_w2 + att_td   (td_w1/td_w2 are [K,N] layouts)
  gemm_bf16_wmma<true, EPI_TANH, false, false><<<g2(BT_, 64), blk, 0, stream>>>(
      Zw, td_w1, tdmid, nullptr, nullptr, nullptr, nullptr, nullptr, BT_, 64, C_);
  gemm_bf16_wmma<false, EPI_NONE, false, false><<<g2(BT_, A_), blk, 0, stream>>>(
      tdmid, td_w2, wraw, att_td, nullptr, nullptr, nullptr, nullptr, BT_, A_, 64);

  wkv_kernel<<<B_ * H_, 64, 0, stream>>>(rbuf, kbuf, vbuf, wraw, att_u, wkvo);

  gn_gate_kernel<<<BT_, blk, 0, stream>>>(wkvo, gbuf, lnx_g, lnx_b, Gg);

  // x2 = x1 + (g*gn) @ wo^T
  gemm_bf16_wmma<false, EPI_NONE, true, false><<<g2(BT_, C_), blk, 0, stream>>>(
      Gg, att_wo, x2, nullptr, x1, nullptr, nullptr, nullptr, BT_, C_, A_);

  // ---- MoE half ----
  ffn_prep_kernel<<<BT_, blk, 0, stream>>>(x2, ln2_g, ln2_b, fmaa_k, fmaa_r, Zk2, Zr2);

  gemm_bf16_wmma<false, EPI_SIGMOID, true, false><<<g2(BT_, C_), blk, 0, stream>>>(
      Zr2, ffn_wr, recept, nullptr, nullptr, nullptr, nullptr, nullptr, BT_, C_, C_);

  // shared expert: val = relu(zk @ sh_wk^T)^2 @ sh_wv^T
  gemm_bf16_wmma<true, EPI_RELUSQ, true, false><<<g2(BT_, I_), blk, 0, stream>>>(
      Zk2, sh_wk, Hk, nullptr, nullptr, nullptr, nullptr, nullptr, BT_, I_, C_);
  gemm_bf16_wmma<false, EPI_NONE, true, false><<<g2(BT_, C_), blk, 0, stream>>>(
      Hk, sh_wv, val, nullptr, nullptr, nullptr, nullptr, nullptr, BT_, C_, I_);

  route_zero_kernel<<<1, 32, 0, stream>>>(counts);
  route_build_kernel<<<(BT_ + 255) / 256, blk, 0, stream>>>(ids, counts, lists);

  for (int e = 0; e < E_; ++e) {
    const int* list_e = lists + (size_t)e * BT_;
    const int* cnt_e = counts + e;
    gemm_bf16_wmma<true, EPI_RELUSQ, true, false><<<g2(BT_, I_), blk, 0, stream>>>(
        Zk2, ex_wk + (size_t)e * I_ * C_, Hk, nullptr, nullptr, list_e, nullptr, cnt_e,
        BT_, I_, C_);
    gemm_bf16_wmma<false, EPI_NONE, true, true><<<g2(BT_, C_), blk, 0, stream>>>(
        Hk, ex_wv + (size_t)e * C_ * I_, val, nullptr, nullptr, nullptr, list_e, cnt_e,
        BT_, C_, I_);
  }

  final_kernel<<<(BT_ * C_ + 255) / 256, blk, 0, stream>>>(x2, recept, val, out);
}
